// SenseVoiceEncoderSmall_73074573574733
// MI455X (gfx1250) — compile-verified, matrix operand order checked by reading, not run on hardware
//
#include <hip/hip_runtime.h>
#include <hip/hip_bf16.h>
#include <math.h>

#define TPB 256

typedef __bf16 v16bf __attribute__((ext_vector_type(16)));
typedef float  v8f   __attribute__((ext_vector_type(8)));

union Frag { unsigned int u[8]; v16bf v; };

__device__ __forceinline__ unsigned int f2bf1(float f) {
  unsigned int u = __float_as_uint(f);
  unsigned int r = u + 0x7FFFu + ((u >> 16) & 1u);   // round-to-nearest-even
  return r >> 16;
}
__device__ __forceinline__ unsigned int packbf(float a, float b) {
  return f2bf1(a) | (f2bf1(b) << 16);
}

// ---------------------------------------------------------------------------
// Generic batched GEMM: C = alpha * (A x B) + bias, optional ReLU.
// A: fp32 [M x K] row-major (lda). B: fp32 [K x N] (ldb) or, if transB,
// [N x K] row-major (ldb = K-stride). Staged as bf16 in LDS, multiplied with
// v_wmma_f32_16x16x32_bf16.  Block tile 64(M) x 128(N), 8 waves = 2x4, each
// wave owns a 32x32 patch (2x2 WMMA tiles).  Host guarantees M%64==0 and
// N%128==0; only a K%32 tail needs zero-padding (the K=560 layer-0 proj).
// Staging is split into a batched load phase (float2 regs) and a pack/store
// phase so all global b64 loads are in flight before the first wait.
// Two-level batch: blockIdx.z -> (b = z/nH, h = z%nH), element strides.
// ---------------------------------------------------------------------------
__global__ __launch_bounds__(256)
void gemm_wmma_bf16(const float* __restrict__ A, const float* __restrict__ Bm,
                    const float* __restrict__ bias, float* __restrict__ C,
                    int M, int N, int K, int lda, int ldb, int ldc,
                    long sAb, long sAh, long sBb, long sBh, long sCb, long sCh,
                    int nH, float alpha, int relu, int transB)
{
  __shared__ unsigned int sA[64 * 16];    //  64 rows x 32 bf16 (packed dwords)
  __shared__ unsigned int sB[128 * 16];   // 128 n-rows x 32 bf16 (N-major)

  const int tid = threadIdx.x;
  const int z  = blockIdx.z;
  const int bb = z / nH, hh = z % nH;
  A  += (long)bb * sAb + (long)hh * sAh;
  Bm += (long)bb * sBb + (long)hh * sBh;
  C  += (long)bb * sCb + (long)hh * sCh;

  const int mBlock = blockIdx.y * 64;
  const int nBlock = blockIdx.x * 128;

  const int wave = tid >> 5;
  const int lane = tid & 31;
  const int hi   = (lane >> 4) & 1;
  const int lo   = lane & 15;
  const int wm   = wave >> 2;     // 0..1  (M)
  const int wn   = wave & 3;      // 0..3  (N)

  v8f zero = {0.f, 0.f, 0.f, 0.f, 0.f, 0.f, 0.f, 0.f};
  v8f acc[2][2];
  acc[0][0] = zero; acc[0][1] = zero; acc[1][0] = zero; acc[1][1] = zero;

  // fragment load + 2x2 WMMA on the currently staged K-slab
  auto mma_step = [&]() {
    Frag fa[2], fb[2];
#pragma unroll
    for (int t = 0; t < 2; ++t) {
      int mrow = wm * 32 + t * 16 + lo;
      int nrow = wn * 32 + t * 16 + lo;
#pragma unroll
      for (int j = 0; j < 8; ++j) {
        // packed-dword column per ISA 16-bit 16x32 layout:
        // j<4 -> j+4*hi ; j>=4 -> 8+(j-4)+4*hi  (two contiguous b128 groups)
        int col = j + 4 * hi + ((j >> 2) << 2);
        fa[t].u[j] = sA[mrow * 16 + col];
        fb[t].u[j] = sB[nrow * 16 + col];
      }
    }
#pragma unroll
    for (int tm = 0; tm < 2; ++tm)
#pragma unroll
      for (int tn = 0; tn < 2; ++tn)
        acc[tm][tn] = __builtin_amdgcn_wmma_f32_16x16x32_bf16(
            false, fa[tm].v, false, fb[tn].v, (short)0, acc[tm][tn],
            false, false);
  };

  const int kFull = K & ~31;
  for (int kb = 0; kb < kFull; kb += 32) {
    // ---- load phase: batch all global b64 loads into registers ----
    float2 aR[4];
#pragma unroll
    for (int u = 0; u < 4; ++u) {
      int i = tid + u * TPB;
      int r = i >> 4, kw = i & 15;
      aR[u] = *(const float2*)(A + (long)(mBlock + r) * lda + kb + 2 * kw);
    }
    if (transB) {
      float2 bR[8];
#pragma unroll
      for (int u = 0; u < 8; ++u) {
        int i = tid + u * TPB;
        int n = i >> 4, kw = i & 15;
        bR[u] = *(const float2*)(Bm + (long)(nBlock + n) * ldb + kb + 2 * kw);
      }
      // ---- pack/store phase ----
#pragma unroll
      for (int u = 0; u < 4; ++u) sA[tid + u * TPB] = packbf(aR[u].x, aR[u].y);
#pragma unroll
      for (int u = 0; u < 8; ++u) sB[tid + u * TPB] = packbf(bR[u].x, bR[u].y);
    } else {
      // K-major weights: 2x2 micro-transpose from two row-pair b64 loads
      float2 b0R[4], b1R[4];
#pragma unroll
      for (int u = 0; u < 4; ++u) {
        int i = tid + u * TPB;
        int np = i >> 4, kw = i & 15;
        int n  = np * 2;
        int k0 = kb + 2 * kw;
        b0R[u] = *(const float2*)(Bm + (long)k0 * ldb + nBlock + n);
        b1R[u] = *(const float2*)(Bm + (long)(k0 + 1) * ldb + nBlock + n);
      }
#pragma unroll
      for (int u = 0; u < 4; ++u) sA[tid + u * TPB] = packbf(aR[u].x, aR[u].y);
#pragma unroll
      for (int u = 0; u < 4; ++u) {
        int i = tid + u * TPB;
        int np = i >> 4, kw = i & 15;
        int n  = np * 2;
        sB[n * 16 + kw]       = packbf(b0R[u].x, b1R[u].x);
        sB[(n + 1) * 16 + kw] = packbf(b0R[u].y, b1R[u].y);
      }
    }
    // --- prefetch next K-slab (global_prefetch_b8), overlaps the MMA work ---
    if (kb + 32 < K) {
      if (tid < 64) {
        __builtin_prefetch(A + (long)(mBlock + tid) * lda + kb + 32, 0, 1);
      } else if (tid < 192) {
        int n = tid - 64;                       // 0..127
        if (transB)
          __builtin_prefetch(Bm + (long)(nBlock + n) * ldb + kb + 32, 0, 1);
        else if (n < 32)
          __builtin_prefetch(Bm + (long)(kb + 32 + n) * ldb + nBlock, 0, 1);
      }
    }
    __syncthreads();
    mma_step();
    __syncthreads();
  }

  if (kFull < K) {                              // K%32 tail (zero-padded)
    for (int i = tid; i < 64 * 16; i += TPB) {
      int r = i >> 4, kw = i & 15;
      int k0 = kFull + 2 * kw;
      const float* ar = A + (long)(mBlock + r) * lda;
      float a0 = (k0     < K) ? ar[k0]     : 0.f;
      float a1 = (k0 + 1 < K) ? ar[k0 + 1] : 0.f;
      sA[i] = packbf(a0, a1);
    }
    for (int i = tid; i < 128 * 16; i += TPB) {
      int n = i >> 4, kw = i & 15;
      int k0 = kFull + 2 * kw;
      float b0 = 0.f, b1 = 0.f;
      if (transB) {
        const float* br = Bm + (long)(nBlock + n) * ldb;
        if (k0     < K) b0 = br[k0];
        if (k0 + 1 < K) b1 = br[k0 + 1];
      } else {
        if (k0     < K) b0 = Bm[(long)k0 * ldb + nBlock + n];
        if (k0 + 1 < K) b1 = Bm[(long)(k0 + 1) * ldb + nBlock + n];
      }
      sB[i] = packbf(b0, b1);
    }
    __syncthreads();
    mma_step();
    __syncthreads();
  }

  // --- epilogue: C/D layout M = r + 8*hi, N = lo ---
#pragma unroll
  for (int tm = 0; tm < 2; ++tm) {
#pragma unroll
    for (int tn = 0; tn < 2; ++tn) {
      int gn = nBlock + wn * 32 + tn * 16 + lo;
      float bv = bias ? bias[gn] : 0.f;
#pragma unroll
      for (int r = 0; r < 8; ++r) {
        int gm = mBlock + wm * 32 + tm * 16 + hi * 8 + r;
        float val = acc[tm][tn][r] * alpha + bv;
        if (relu) val = fmaxf(val, 0.f);
        C[(long)gm * ldc + gn] = val;
      }
    }
  }
}

// ---------------------------------------------------------------------------
// LayerNorm over last dim (one block per row), optional mask multiply.
// ---------------------------------------------------------------------------
__global__ __launch_bounds__(256)
void ln_kernel(const float* __restrict__ x, const float* __restrict__ w,
               const float* __restrict__ b, float* __restrict__ out,
               int D, const float* __restrict__ mask)
{
  __shared__ float red[TPB];
  const long row = blockIdx.x;
  const float* xr = x + row * D;
  float s = 0.f;
  for (int i = threadIdx.x; i < D; i += TPB) s += xr[i];
  red[threadIdx.x] = s; __syncthreads();
  for (int st = TPB / 2; st > 0; st >>= 1) {
    if (threadIdx.x < st) red[threadIdx.x] += red[threadIdx.x + st];
    __syncthreads();
  }
  float mean = red[0] / (float)D;
  __syncthreads();
  float v = 0.f;
  for (int i = threadIdx.x; i < D; i += TPB) { float d = xr[i] - mean; v += d * d; }
  red[threadIdx.x] = v; __syncthreads();
  for (int st = TPB / 2; st > 0; st >>= 1) {
    if (threadIdx.x < st) red[threadIdx.x] += red[threadIdx.x + st];
    __syncthreads();
  }
  float rs = rsqrtf(red[0] / (float)D + 1e-5f);
  float mk = mask ? mask[row] : 1.f;
  float* orow = out + row * D;
  for (int i = threadIdx.x; i < D; i += TPB)
    orow[i] = ((xr[i] - mean) * rs * w[i] + b[i]) * mk;
}

// ---------------------------------------------------------------------------
// Row softmax with additive mask term (mask-1)*1e4; one block per row.
// row index covers (b, h, q); b = row / (H*T).
// ---------------------------------------------------------------------------
__global__ __launch_bounds__(256)
void softmax_kernel(float* __restrict__ sc, const float* __restrict__ mask,
                    int T, int HT)
{
  __shared__ float red[TPB];
  const long row = blockIdx.x;
  const int  b   = (int)(row / HT);
  float* sr = sc + row * T;
  const float* mr = mask + (long)b * T;

  float mx = -3.0e38f;
  for (int i = threadIdx.x; i < T; i += TPB) {
    float v = sr[i] + (mr[i] - 1.f) * 10000.f;
    sr[i] = v;
    mx = fmaxf(mx, v);
  }
  red[threadIdx.x] = mx; __syncthreads();
  for (int st = TPB / 2; st > 0; st >>= 1) {
    if (threadIdx.x < st) red[threadIdx.x] = fmaxf(red[threadIdx.x], red[threadIdx.x + st]);
    __syncthreads();
  }
  mx = red[0]; __syncthreads();
  float sum = 0.f;
  for (int i = threadIdx.x; i < T; i += TPB) {
    float e = __expf(sr[i] - mx);
    sr[i] = e;
    sum += e;
  }
  red[threadIdx.x] = sum; __syncthreads();
  for (int st = TPB / 2; st > 0; st >>= 1) {
    if (threadIdx.x < st) red[threadIdx.x] += red[threadIdx.x + st];
    __syncthreads();
  }
  float inv = 1.f / red[0];
  for (int i = threadIdx.x; i < T; i += TPB) sr[i] *= inv;
}

// ---------------------------------------------------------------------------
// FSMN: depthwise conv (K=11, SAME pad 5/5) over vm = v*mask, plus vm residual.
// v lives inside qkv at column offset voff with row stride ldq.
// ---------------------------------------------------------------------------
__global__ __launch_bounds__(256)
void fsmn_kernel(const float* __restrict__ qkv, const float* __restrict__ mask,
                 const float* __restrict__ w, float* __restrict__ out,
                 long total, int T, int D, int ldq, int voff)
{
  long idx = (long)blockIdx.x * TPB + threadIdx.x;
  if (idx >= total) return;
  int  d  = (int)(idx % D);
  long bt = idx / D;
  int  t  = (int)(bt % T);
  long b  = bt / T;
  const float* vb = qkv + voff + d;
  float acc = vb[bt * ldq] * mask[bt];      // residual vm
#pragma unroll
  for (int j = 0; j < 11; ++j) {
    int tt = t + j - 5;
    if (tt >= 0 && tt < T) {
      long bt2 = b * T + tt;
      acc += w[d * 11 + j] * vb[bt2 * ldq] * mask[bt2];
    }
  }
  out[bt * D + d] = acc;
}

__global__ __launch_bounds__(256)
void add3_kernel(float* __restrict__ x, const float* __restrict__ a,
                 const float* __restrict__ b, long n)
{
  long i = (long)blockIdx.x * TPB + threadIdx.x;
  if (i < n) x[i] += a[i] + b[i];
}

__global__ __launch_bounds__(256)
void add2_kernel(float* __restrict__ o, const float* __restrict__ a,
                 const float* __restrict__ b, long n)
{
  long i = (long)blockIdx.x * TPB + threadIdx.x;
  if (i < n) o[i] = a[i] + b[i];
}

// h = x*sqrt(512) + sinusoidal PE (positions start at 1)
__global__ __launch_bounds__(256)
void pe_kernel(const float* __restrict__ x, float* __restrict__ h,
               int T, int Din, long n)
{
  long idx = (long)blockIdx.x * TPB + threadIdx.x;
  if (idx >= n) return;
  int  i  = (int)(idx % Din);
  long bt = idx / Din;
  int  t  = (int)(bt % T);
  int half = Din >> 1;
  float logts = __logf(10000.f) / (float)(half - 1);
  float pos = (float)(t + 1);
  float val = (i < half)
      ? __sinf(pos * __expf(-logts * (float)i))
      : __cosf(pos * __expf(-logts * (float)(i - half)));
  h[idx] = x[idx] * 22.62741699796952f + val;
}

// ---------------------------------------------------------------------------
extern "C" void kernel_launch(void* const* d_in, const int* in_sizes, int n_in,
                              void* d_out, int out_size, void* d_ws, size_t ws_size,
                              hipStream_t stream) {
  (void)in_sizes; (void)n_in; (void)out_size; (void)ws_size;
  const int Bsz = 8, T = 512, D = 512, H = 4, FF = 2048, DIN = 560;
  const long BT = (long)Bsz * T;           // 4096
  const long n512 = BT * D;

  // ---- inputs (setup_inputs dict insertion order, nested dict order) ----
  const float* x_in  = (const float*)d_in[0];
  const float* mask  = (const float*)d_in[1];
  const float* p0_n1w  = (const float*)d_in[2];
  const float* p0_n1b  = (const float*)d_in[3];
  const float* p0_wqkv = (const float*)d_in[4];
  const float* p0_bqkv = (const float*)d_in[5];
  const float* p0_wout = (const float*)d_in[6];
  const float* p0_bout = (const float*)d_in[7];
  const float* p0_fsmn = (const float*)d_in[8];
  const int ENC = 9, TP = 22;              // stacked-param base indices
  const float* an_w = (const float*)d_in[35];
  const float* an_b = (const float*)d_in[36];
  const float* tn_w = (const float*)d_in[37];
  const float* tn_b = (const float*)d_in[38];

  // ---- workspace carve-up ----
  float* ws = (float*)d_ws;
  long off = 0;
  auto carve = [&](long n) { float* p = ws + off; off += n; return p; };
  float* xA     = carve(BT * D);
  float* xB     = carve(BT * D);
  float* xn     = carve(BT * DIN);
  float* qkv    = carve(BT * 3 * D);
  float* fsm    = carve(BT * D);
  float* scores = carve((long)Bsz * H * T * T);
  float* att    = carve(BT * D);
  float* attout = carve(BT * D);
  float* ff1    = carve(BT * FF);
  float* h560   = ff1;                     // alias: only used before layer stack
  float* ff2    = att;                     // alias: att is free after out-proj

  dim3 blk(TPB);
  auto gemm = [&](const float* A, int lda, const float* Bm, int ldb,
                  const float* bias, float* C, int ldc, int M, int N, int K,
                  float alpha, int relu, int transB,
                  int nB = 1, int nH = 1,
                  long sAb = 0, long sAh = 0, long sBb = 0, long sBh = 0,
                  long sCb = 0, long sCh = 0) {
    dim3 g((N + 127) / 128, (M + 63) / 64, nB * nH);
    gemm_wmma_bf16<<<g, blk, 0, stream>>>(A, Bm, bias, C, M, N, K, lda, ldb,
        ldc, sAb, sAh, sBb, sBh, sCb, sCh, nH, alpha, relu, transB);
  };

  auto run_attn = [&](const float* curx, const float* n1w, const float* n1b,
                      const float* wq, const float* bq,
                      const float* wo, const float* bo,
                      const float* fw, int Din) {
    ln_kernel<<<(int)BT, blk, 0, stream>>>(curx, n1w, n1b, xn, Din, nullptr);
    gemm(xn, Din, wq, 3 * D, bq, qkv, 3 * D, (int)BT, 3 * D, Din, 1.f, 0, 0);
    fsmn_kernel<<<(int)((BT * D + TPB - 1) / TPB), blk, 0, stream>>>(
        qkv, mask, fw, fsm, BT * D, T, D, 3 * D, 2 * D);
    // scores = (Q * DK^-0.5) . K^T   per (b,h)
    gemm(qkv, 3 * D, qkv + D, 3 * D, nullptr, scores, T, T, T, 128,
         0.08838834764831845f, 0, 1, Bsz, H,
         (long)T * 3 * D, 128, (long)T * 3 * D, 128,
         (long)H * T * T, (long)T * T);
    softmax_kernel<<<Bsz * H * T, blk, 0, stream>>>(scores, mask, T, H * T);
    // att = P . V   per (b,h), written into [B,T, h*128 + dk]
    gemm(scores, T, qkv + 2 * D, 3 * D, nullptr, att, D, T, 128, T,
         1.f, 0, 0, Bsz, H,
         (long)H * T * T, (long)T * T, (long)T * 3 * D, 128,
         (long)T * D, 128);
    gemm(att, D, wo, D, bo, attout, D, (int)BT, D, D, 1.f, 0, 0);
  };

  // ---- front end: PE + layer 0 (no residual / FFN) ----
  const long npe = BT * DIN;
  pe_kernel<<<(int)((npe + TPB - 1) / TPB), blk, 0, stream>>>(x_in, h560, T, DIN, npe);
  run_attn(h560, p0_n1w, p0_n1b, p0_wqkv, p0_bqkv, p0_wout, p0_bout, p0_fsmn, DIN);
  add2_kernel<<<(int)((n512 + TPB - 1) / TPB), blk, 0, stream>>>(xA, attout, fsm, n512);
  float* cur = xA;

  struct SP { const float *n1w,*n1b,*wq,*bq,*wo,*bo,*fw,*n2w,*n2b,*w1,*b1,*w2,*b2; };
  auto params = [&](int base, int l) {
    SP p;
    p.n1w = (const float*)d_in[base + 0] + (long)l * D;
    p.n1b = (const float*)d_in[base + 1] + (long)l * D;
    p.wq  = (const float*)d_in[base + 2] + (long)l * D * 3 * D;
    p.bq  = (const float*)d_in[base + 3] + (long)l * 3 * D;
    p.wo  = (const float*)d_in[base + 4] + (long)l * D * D;
    p.bo  = (const float*)d_in[base + 5] + (long)l * D;
    p.fw  = (const float*)d_in[base + 6] + (long)l * D * 11;
    p.n2w = (const float*)d_in[base + 7] + (long)l * D;
    p.n2b = (const float*)d_in[base + 8] + (long)l * D;
    p.w1  = (const float*)d_in[base + 9] + (long)l * D * FF;
    p.b1  = (const float*)d_in[base + 10] + (long)l * FF;
    p.w2  = (const float*)d_in[base + 11] + (long)l * FF * D;
    p.b2  = (const float*)d_in[base + 12] + (long)l * D;
    return p;
  };

  auto full_layer = [&](const SP& p) {
    run_attn(cur, p.n1w, p.n1b, p.wq, p.bq, p.wo, p.bo, p.fw, D);
    add3_kernel<<<(int)((n512 + TPB - 1) / TPB), blk, 0, stream>>>(cur, attout, fsm, n512);
    ln_kernel<<<(int)BT, blk, 0, stream>>>(cur, p.n2w, p.n2b, xn, D, nullptr);
    gemm(xn, D, p.w1, FF, p.b1, ff1, FF, (int)BT, FF, D, 1.f, 1, 0);   // ReLU
    gemm(ff1, FF, p.w2, D, p.b2, ff2, D, (int)BT, D, FF, 1.f, 0, 0);
    add2_kernel<<<(int)((n512 + TPB - 1) / TPB), blk, 0, stream>>>(cur, cur, ff2, n512);
  };

  for (int l = 0; l < 49; ++l) full_layer(params(ENC, l));
  ln_kernel<<<(int)BT, blk, 0, stream>>>(cur, an_w, an_b, xB, D, mask);
  cur = xB;
  for (int l = 0; l < 20; ++l) full_layer(params(TP, l));
  ln_kernel<<<(int)BT, blk, 0, stream>>>(cur, tn_w, tn_b, (float*)d_out, D, mask);
}
